// modified_net2_46291157517018
// MI455X (gfx1250) — matrix operand order repaired and checked by reading, not compiled, not run on hardware
//
#include <hip/hip_runtime.h>
#include <hip/hip_bf16.h>

// ---------------------------------------------------------------------------
// MI455X (gfx1250) SS3D / VSS network, round 3.
//  - k_gemm32: 4 M-tiles per wave -> 4 independent v_wmma per wave, B-frag
//    loaded once, no inter-WMMA hazard stalls.
//  - k_scan: double-buffered LDS staging via global_load_async_to_lds_b128
//    (ASYNCcnt) so the serial recurrence reads LDS, not L2.
// ---------------------------------------------------------------------------

typedef __attribute__((ext_vector_type(16))) _Float16 v16h;
typedef __attribute__((ext_vector_type(2)))  _Float16 h2;
typedef __attribute__((ext_vector_type(8)))  float    v8f;

#define DEV static __device__ __forceinline__

DEV v8f wmma16(v16h a, v16h b, v8f c) {
  return __builtin_amdgcn_wmma_f32_16x16x32_f16(false, a, false, b, (short)0, c,
                                                false, false);
}

DEV float actf(float x, int act) {
  switch (act) {
    case 1: return fmaxf(x, 0.f);
    case 2: return x / (1.f + __expf(-x));                       // silu
    case 3: return 0.5f * x * (1.f + erff(x * 0.70710678118f));  // gelu (exact)
    case 4: return 1.f / (1.f + __expf(-x));                     // sigmoid
  }
  return x;
}

// async global -> LDS copy, 16 bytes per lane (gfx1250, tracked by ASYNCcnt)
DEV void async_cp_b128(const float* gsrc, const float* lds_dst) {
  unsigned ldsa = (unsigned)(unsigned long long)(const void*)lds_dst;
  asm volatile("global_load_async_to_lds_b128 %0, %1, off"
               :: "v"(ldsa), "v"(gsrc) : "memory");
}

// ---------------------------------------------------------------------------
// Specialized WMMA GEMM:  C[p,n] = sum_{k<32} A[p,k]*W[n,k] (+bias) (+resid)
// p in [0,16384) = (p>>12, p&4095).  A K-contiguous.  One wave = 4 M-tiles x
// one 16-wide N-tile: B loaded once, 4 independent v_wmma.
// ---------------------------------------------------------------------------
template <bool NGUARD, bool RES>
__global__ void k_gemm32(const float* __restrict__ A, const float* __restrict__ W,
                         const float* __restrict__ bias,
                         const float* __restrict__ resid,
                         const float* __restrict__ skipw,
                         float* __restrict__ C, int N,
                         long a_base, long a_sout, long a_sin,
                         long w_sn,
                         long c_base, long c_sout, long c_sin, long c_sn,
                         long r_base, long r_sout, long r_sin) {
  int ntn  = (N + 15) >> 4;
  int mt4  = blockIdx.x / ntn, nt = blockIdx.x % ntn;
  int lane = threadIdx.x & 31;
  int hp   = lane >> 4;
  int n    = nt * 16 + (lane & 15);

  int nc = NGUARD ? (n < N ? n : N - 1) : n;   // clamp: junk cols never stored
  const float* wp = W + (long)nc * w_sn;
  v16h bf;
#pragma unroll
  for (int j = 0; j < 8; ++j) {
    float2 v = *(const float2*)(wp + hp * 16 + 2 * j);
    bf[2 * j]     = (_Float16)v.x;
    bf[2 * j + 1] = (_Float16)v.y;
  }

  v16h af[4];
#pragma unroll
  for (int i = 0; i < 4; ++i) {
    int m = (mt4 * 4 + i) * 16 + (lane & 15);
    const float* ap = A + a_base + (long)(m >> 12) * a_sout + (long)(m & 4095) * a_sin;
#pragma unroll
    for (int j = 0; j < 8; ++j) {         // A frag (contiguous k = 16*hp + s, matches B)
      int kp = hp * 16 + 2 * j;
      float2 v = *(const float2*)(ap + kp);
      af[i][2 * j]     = (_Float16)v.x;
      af[i][2 * j + 1] = (_Float16)v.y;
    }
  }

  v8f acc[4];
#pragma unroll
  for (int i = 0; i < 4; ++i) {           // independent accs: no hazard NOPs
    v8f zero = {};
    acc[i] = wmma16(af[i], bf, zero);
  }

  if (NGUARD && n >= N) return;
  float bv = bias ? bias[n] : 0.f;
#pragma unroll
  for (int i = 0; i < 4; ++i) {
#pragma unroll
    for (int v = 0; v < 8; ++v) {
      int  mm = (mt4 * 4 + i) * 16 + v + 8 * hp;  // D layout: VGPR v -> row v+8*hp
      long qo = mm >> 12, qi = mm & 4095;
      long co = c_base + qo * c_sout + qi * c_sin + (long)n * c_sn;
      float val = acc[i][v] + bv;
      if (RES) {
        long ro = r_base + qo * r_sout + qi * r_sin + n;   // resid chan-last
        val += resid[ro] * (skipw ? skipw[n] : 1.f);
      }
      C[co] = val;
    }
  }
}

// ---------------------------------------------------------------------------
// Repack conv weights (Co,Ci,3,3,3) f32 -> f16 [tap][co][ci] for B-fragments.
// ---------------------------------------------------------------------------
__global__ void k_repack_w(const float* __restrict__ w, _Float16* __restrict__ wf) {
  int idx = blockIdx.x * blockDim.x + threadIdx.x;   // 27*32*32 = 27648
  if (idx >= 27648) return;
  int ci = idx & 31, co = (idx >> 5) & 31, t = idx >> 10;
  wf[idx] = (_Float16)w[(co * 32 + ci) * 27 + t];
}

// ---------------------------------------------------------------------------
// Implicit-GEMM 3x3x3 conv, Cin=Cout=32, pad=1, NDHWC, Wo==16.
// One wave per output row; 27 taps x 2 N-tiles of v_wmma.
// ---------------------------------------------------------------------------
__global__ void k_conv32_wmma(const float* __restrict__ in,
                              const _Float16* __restrict__ wf,
                              const float* __restrict__ bias,
                              float* __restrict__ out,
                              int Di, int Hi, int Wi,
                              int Do, int Ho, int Wo, int stride) {
  int rid = blockIdx.x;
  int y = rid % Ho;
  int z = (rid / Ho) % Do;
  int b = rid / (Ho * Do);
  int lane = threadIdx.x & 31;
  int mloc = lane & 15;   // output x
  int hp   = lane >> 4;
  int nn   = lane & 15;   // output channel in tile

  int  xi0 = mloc * stride - 1;
  v8f acc0 = {}, acc1 = {};

  for (int t = 0; t < 27; ++t) {
    int dz = t / 9, dy = (t / 3) % 3, dx = t % 3;
    int zi = z * stride + dz - 1;
    int yi = y * stride + dy - 1;
    if (zi < 0 || zi >= Di || yi < 0 || yi >= Hi) continue;  // uniform branch
    int  xi  = xi0 + dx;
    bool xok = (xi >= 0 && xi < Wi);
    int  xic = xi < 0 ? 0 : (xi >= Wi ? Wi - 1 : xi);        // clamp: safe addr
    const float* ap = in + ((((long)b * Di + zi) * Hi + yi) * Wi + xic) * 32;

    v16h af;
#pragma unroll
    for (int j = 0; j < 8; ++j) {
      int kp = hp * 16 + 2 * j;
      float2 v = *(const float2*)(ap + kp);
      af[2 * j]     = (_Float16)(xok ? v.x : 0.f);
      af[2 * j + 1] = (_Float16)(xok ? v.y : 0.f);
    }
    const _Float16* w0 = wf + ((long)t * 32 + nn) * 32;
    const _Float16* w1 = w0 + 16 * 32;
    v16h bf0, bf1;
#pragma unroll
    for (int j = 0; j < 8; ++j) {
      int ci = hp * 16 + 2 * j;
      h2 p0 = *(const h2*)(w0 + ci);
      h2 p1 = *(const h2*)(w1 + ci);
      bf0[2 * j] = p0[0]; bf0[2 * j + 1] = p0[1];
      bf1[2 * j] = p1[0]; bf1[2 * j + 1] = p1[1];
    }
    acc0 = wmma16(af, bf0, acc0);
    acc1 = wmma16(af, bf1, acc1);
  }

  long ob = (((long)b * Do + z) * Ho + y) * Wo * 32;
  float b0 = bias[nn], b1 = bias[nn + 16];
#pragma unroll
  for (int v = 0; v < 8; ++v) {
    int x = v + 8 * hp;
    out[ob + (long)x * 32 + nn]      = acc0[v] + b0;
    out[ob + (long)x * 32 + nn + 16] = acc1[v] + b1;
  }
}

// ---------------------------------------------------------------------------
// Expand conv 1->32, NDHWC output.  Taps broadcast across the 32 c-lanes.
// ---------------------------------------------------------------------------
__global__ void k_expand(const float* __restrict__ x, const float* __restrict__ w,
                         const float* __restrict__ bias, float* __restrict__ out,
                         long total) {
  long idx = (long)blockIdx.x * blockDim.x + threadIdx.x;
  if (idx >= total) return;
  int c = idx & 31;
  long r = idx >> 5;
  int s = (int)(r & 32767);
  int b = (int)(r >> 15);
  int zz = s >> 10, yy = (s >> 5) & 31, xx = s & 31;
  const float* xb = x + (long)b * 32768;
  float acc = bias[c];
  for (int t = 0; t < 27; ++t) {
    int zi = zz + t / 9 - 1, yi = yy + (t / 3) % 3 - 1, xi = xx + t % 3 - 1;
    if (zi < 0 || zi >= 32 || yi < 0 || yi >= 32 || xi < 0 || xi >= 32) continue;
    acc += xb[(zi * 32 + yi) * 32 + xi] * w[c * 27 + t];
  }
  out[idx] = acc;
}

// ---------------------------------------------------------------------------
// GroupNorm(groups=8, C=32) on NDHWC, float4 per position, optional ReLU.
// ---------------------------------------------------------------------------
__global__ void k_gn_cl(const float* __restrict__ in, float* __restrict__ out,
                        const float* __restrict__ gw, const float* __restrict__ gb,
                        int sp, int relu) {
  int b = blockIdx.x >> 3, gi = blockIdx.x & 7;
  const float* base = in + (long)b * sp * 32 + gi * 4;
  float s0 = 0.f, s1 = 0.f;
  for (int s = threadIdx.x; s < sp; s += 256) {
    float4 v = *(const float4*)(base + (long)s * 32);
    s0 += v.x + v.y + v.z + v.w;
    s1 += v.x * v.x + v.y * v.y + v.z * v.z + v.w * v.w;
  }
  __shared__ float r0[256], r1[256];
  r0[threadIdx.x] = s0; r1[threadIdx.x] = s1;
  __syncthreads();
  for (int st = 128; st > 0; st >>= 1) {
    if (threadIdx.x < st) {
      r0[threadIdx.x] += r0[threadIdx.x + st];
      r1[threadIdx.x] += r1[threadIdx.x + st];
    }
    __syncthreads();
  }
  float Nt  = (float)sp * 4.f;
  float m   = r0[0] / Nt;
  float var = r1[0] / Nt - m * m;
  float inv = rsqrtf(var + 1e-5f);
  float4 g4 = *(const float4*)(gw + gi * 4);
  float4 b4 = *(const float4*)(gb + gi * 4);
  float* ob = out + (long)b * sp * 32 + gi * 4;
  for (int s = threadIdx.x; s < sp; s += 256) {
    float4 v = *(const float4*)(base + (long)s * 32);
    float4 o;
    o.x = (v.x - m) * inv * g4.x + b4.x;
    o.y = (v.y - m) * inv * g4.y + b4.y;
    o.z = (v.z - m) * inv * g4.z + b4.z;
    o.w = (v.w - m) * inv * g4.w + b4.w;
    if (relu) { o.x = fmaxf(o.x, 0.f); o.y = fmaxf(o.y, 0.f);
                o.z = fmaxf(o.z, 0.f); o.w = fmaxf(o.w, 0.f); }
    *(float4*)(ob + (long)s * 32) = o;
  }
}

// ---------------------------------------------------------------------------
// LayerNorm over C=32: one wave32 per position, full-wave shuffle reductions.
// Optional fused *silu(z), optional channel-first output (for CAB convs).
// ---------------------------------------------------------------------------
__global__ void k_ln32(const float* __restrict__ in, float* __restrict__ out,
                       const float* __restrict__ g, const float* __restrict__ bta,
                       float eps, const float* __restrict__ z, int zstride,
                       int cf_out, int sp, int P) {
  int wid  = (int)((blockIdx.x * (long)blockDim.x + threadIdx.x) >> 5);
  int lane = threadIdx.x & 31;
  if (wid >= P) return;
  float v = in[(long)wid * 32 + lane];
  float m = v;
#pragma unroll
  for (int o = 16; o >= 1; o >>= 1) m += __shfl_xor(m, o, 32);
  m *= (1.f / 32.f);
  float d  = v - m;
  float vv = d * d;
#pragma unroll
  for (int o = 16; o >= 1; o >>= 1) vv += __shfl_xor(vv, o, 32);
  vv *= (1.f / 32.f);
  float xn = d * rsqrtf(vv + eps) * g[lane] + bta[lane];
  if (z) {
    float zz = z[(long)wid * zstride + lane];
    xn *= zz / (1.f + __expf(-zz));
  }
  if (cf_out) {
    int b = wid / sp, s = wid % sp;
    out[((long)b * 32 + lane) * sp + s] = xn;
  } else {
    out[(long)wid * 32 + lane] = xn;
  }
}

// ---------------------------------------------------------------------------
// Selective scan with double-buffered async LDS staging.
// grid = B*K (32 blocks), block = 512 (= C*N).  Lane = one state (c,n).
// Per 128-step chunk: delta(32x128), u(128x32), B(16x128), C(16x128) staged
// via global_load_async_to_lds_b128 (6 per thread), rows padded to 132 floats
// so the n-indexed LDS reads are bank-conflict-free.
// ---------------------------------------------------------------------------
#define SCAN_BUF 12544   // 32*132 + 16*132 + 16*132 + 128*32
#define OFF_B    4224
#define OFF_C    6336
#define OFF_U    8448

DEV void scan_stage(float* buf, const float* __restrict__ delta,
                    const float* __restrict__ xs, const float* __restrict__ xdbl,
                    long bk, int l0, int t) {
  {  // delta rows: 16 threads x 8 floats per row
    int row = t >> 4, col = (t & 15) * 8;
    const float* src = delta + (bk * 32 + row) * 4096 + l0 + col;
    float* dst = buf + row * 132 + col;
    async_cp_b128(src, dst);
    async_cp_b128(src + 4, dst + 4);
  }
  {  // u: contiguous chunk of [bk,l,c]
    const float* src = xs + (bk * 4096 + l0) * 32 + t * 8;
    float* dst = buf + OFF_U + t * 8;
    async_cp_b128(src, dst);
    async_cp_b128(src + 4, dst + 4);
  }
  {  // B rows: 32 threads x 4 floats per row
    int row = t >> 5, col = (t & 31) * 4;
    async_cp_b128(xdbl + (bk * 34 + 2 + row) * 4096 + l0 + col,
                  buf + OFF_B + row * 132 + col);
    async_cp_b128(xdbl + (bk * 34 + 18 + row) * 4096 + l0 + col,
                  buf + OFF_C + row * 132 + col);
  }
}

__global__ void k_scan(const float* __restrict__ delta, const float* __restrict__ xs,
                       const float* __restrict__ xdbl, const float* __restrict__ Alog,
                       const float* __restrict__ Ds, float* __restrict__ ys) {
  __shared__ float smem[2 * SCAN_BUF];
  int b = blockIdx.x >> 3;
  int k = blockIdx.x & 7;
  int t = threadIdx.x;
  int c = t >> 4;
  int n = t & 15;
  long bk = (long)b * 8 + k;
  float Ac = -__expf(Alog[(k * 32 + c) * 16 + n]);
  float Dv = Ds[k * 32 + c];
  float* yp = ys + (bk * 32 + c) * 4096;
  float h = 0.f;

  scan_stage(smem, delta, xs, xdbl, bk, 0, t);
  for (int ch = 0; ch < 32; ++ch) {
    int cur = ch & 1;
    if (ch + 1 < 32) {
      scan_stage(smem + (cur ^ 1) * SCAN_BUF, delta, xs, xdbl, bk,
                 (ch + 1) * 128, t);
      asm volatile("s_wait_asynccnt 6" ::: "memory");  // this chunk landed
    } else {
      asm volatile("s_wait_asynccnt 0" ::: "memory");
    }
    __syncthreads();

    const float* buf = smem + cur * SCAN_BUF;
    const float* Sd = buf + c * 132;
    const float* Su = buf + OFF_U + c;
    const float* Sb = buf + OFF_B + n * 132;
    const float* Sc = buf + OFF_C + n * 132;
    float* ypo = yp + ch * 128;
    for (int j = 0; j < 128; ++j) {
      float d = Sd[j], u = Su[j * 32], Bv = Sb[j], Cv = Sc[j];
      h = __expf(d * Ac) * h + d * u * Bv;
      float yv = h * Cv;
#pragma unroll
      for (int o = 8; o >= 1; o >>= 1) yv += __shfl_xor(yv, o, 16);
      if (n == 0) ypo[j] = yv + Dv * u;
    }
    __syncthreads();   // everyone done reading buf before it is re-staged
  }
}

// ---------------------------------------------------------------------------
// delta = softplus(dtr . dt_w + dt_b), DT_RANK = 2, layout [bk,c,l].
// ---------------------------------------------------------------------------
__global__ void k_dt(const float* __restrict__ xdbl, const float* __restrict__ dtw,
                     const float* __restrict__ dtb, float* __restrict__ delta,
                     long total) {
  long idx = (long)blockIdx.x * blockDim.x + threadIdx.x;
  if (idx >= total) return;
  int l = idx & 4095;
  long t = idx >> 12;
  int c = t & 31; t >>= 5;
  int k = t & 7;
  int b = (int)(t >> 3);
  long bk = (long)b * 8 + k;
  float d0 = xdbl[(bk * 34 + 0) * 4096 + l];
  float d1 = xdbl[(bk * 34 + 1) * 4096 + l];
  float v  = d0 * dtw[(k * 32 + c) * 2] + d1 * dtw[(k * 32 + c) * 2 + 1] + dtb[k * 32 + c];
  delta[idx] = (v > 20.f) ? v : log1pf(__expf(v));
}

// ---------------------------------------------------------------------------
// Depthwise 3x3x3 conv + SiLU, reading x-half of xz (stride-64 chan-last).
// ---------------------------------------------------------------------------
__global__ void k_dwconv_silu(const float* __restrict__ xz, const float* __restrict__ w,
                              const float* __restrict__ bias, float* __restrict__ xc,
                              long total) {
  long idx = (long)blockIdx.x * blockDim.x + threadIdx.x;
  if (idx >= total) return;
  int c = idx & 31;
  int s = (int)((idx >> 5) & 4095);
  int b = (int)(idx >> 17);
  int zz = s >> 8, yy = (s >> 4) & 15, xx = s & 15;
  const float* zb = xz + (long)b * 4096 * 64 + c;
  float acc = bias[c];
  for (int t = 0; t < 27; ++t) {
    int zi = zz + t / 9 - 1, yi = yy + (t / 3) % 3 - 1, xi = xx + t % 3 - 1;
    if (zi < 0 || zi >= 16 || yi < 0 || yi >= 16 || xi < 0 || xi >= 16) continue;
    acc += zb[(long)((zi * 16 + yi) * 16 + xi) * 64] * w[c * 27 + t];
  }
  xc[idx] = acc / (1.f + __expf(-acc));
}

// ---------------------------------------------------------------------------
// Build xs[b,k,l,c] (chan-last per direction) from xc[b,s,c].
// ---------------------------------------------------------------------------
__global__ void k_gather(const float* __restrict__ xc, float* __restrict__ xs,
                         long total) {
  long idx = (long)blockIdx.x * blockDim.x + threadIdx.x;
  if (idx >= total) return;
  int c = idx & 31;
  int l = (int)((idx >> 5) & 4095);
  int k = (int)((idx >> 17) & 7);
  int b = (int)(idx >> 20);
  int i = k >> 1, rev = k & 1;
  int ll = rev ? (4095 - l) : l;
  int i0 = ll >> 8, i1 = (ll >> 4) & 15, i2 = ll & 15;
  int z, y, x;
  switch (i) {
    case 0: z = i0; y = i1; x = i2; break;  // (D,H,W)
    case 1: y = i0; x = i1; z = i2; break;  // (H,W,D)
    case 2: x = i0; z = i1; y = i2; break;  // (W,D,H)
    default: x = i0; y = i1; z = i2; break; // (W,H,D)
  }
  xs[idx] = xc[((long)b * 4096 + ((z * 16 + y) * 16 + x)) * 32 + c];
}

// ---------------------------------------------------------------------------
// Merge 8 direction outputs (ys layout [bk,c,l]) into channel-last (b,s,c).
// ---------------------------------------------------------------------------
__global__ void k_merge(const float* __restrict__ ys, float* __restrict__ ycl,
                        long total) {
  long idx = (long)blockIdx.x * blockDim.x + threadIdx.x;
  if (idx >= total) return;
  int c = idx & 31;
  int s = (int)((idx >> 5) & 4095);
  int b = (int)(idx >> 17);
  int z = s >> 8, y = (s >> 4) & 15, x = s & 15;
  int ls[4];
  ls[0] = (z * 16 + y) * 16 + x;
  ls[1] = (y * 16 + x) * 16 + z;
  ls[2] = (x * 16 + z) * 16 + y;
  ls[3] = (x * 16 + y) * 16 + z;
  float sum = 0.f;
#pragma unroll
  for (int i = 0; i < 4; ++i) {
    long base = (((long)b * 8 + 2 * i) * 32 + c) * 4096;
    sum += ys[base + ls[i]] + ys[base + 32L * 4096 + (4095 - ls[i])];
  }
  ycl[idx] = sum;
}

// --------------------------- layout shuffle --------------------------------
__global__ void k_cl_to_cf(const float* __restrict__ src, float* __restrict__ dst,
                           int Ctot, int coff, long total) {
  long idx = (long)blockIdx.x * blockDim.x + threadIdx.x;
  if (idx >= total) return;
  int c = idx & 31;
  long s = (idx >> 5) & 4095;
  int b = (int)(idx >> 17);
  dst[((long)b * Ctot + coff + c) * 4096 + s] = src[idx];
}

// ---------------------------------------------------------------------------
// Direct conv3d (channel-first; CAB 32->10->32 convs), fused activation.
// ---------------------------------------------------------------------------
__global__ void k_conv_direct(const float* __restrict__ in, const float* __restrict__ w,
                              const float* __restrict__ bias, float* __restrict__ out,
                              int Ci, int Co, int act, long total) {
  long idx = (long)blockIdx.x * blockDim.x + threadIdx.x;
  if (idx >= total) return;
  long r = idx;
  int x = r % 16; r /= 16;
  int y = r % 16; r /= 16;
  int zc = r % 16; r /= 16;
  int co = r % Co;
  int b  = (int)(r / Co);
  float acc = bias[co];
  for (int ci = 0; ci < Ci; ++ci) {
    const float* ip = in + ((long)b * Ci + ci) * 4096;
    const float* wp = w + ((long)co * Ci + ci) * 27;
    for (int t = 0; t < 27; ++t) {
      int zi = zc + t / 9 - 1, yi = y + (t / 3) % 3 - 1, xi = x + t % 3 - 1;
      if (zi < 0 || zi >= 16 || yi < 0 || yi >= 16 || xi < 0 || xi >= 16) continue;
      acc += ip[(zi * 16 + yi) * 16 + xi] * wp[t];
    }
  }
  out[idx] = actf(acc, act);
}

// ----------------------------- CAB pieces ----------------------------------
__global__ void k_pool(const float* __restrict__ h, float* __restrict__ a, int sp) {
  int bc = blockIdx.x;
  __shared__ float r[256];
  float s = 0.f;
  for (int i = threadIdx.x; i < sp; i += 256) s += h[(long)bc * sp + i];
  r[threadIdx.x] = s;
  __syncthreads();
  for (int st = 128; st > 0; st >>= 1) {
    if (threadIdx.x < st) r[threadIdx.x] += r[threadIdx.x + st];
    __syncthreads();
  }
  if (threadIdx.x == 0) a[bc] = r[0] / sp;
}

__global__ void k_camlp(const float* __restrict__ a0, const float* __restrict__ w1,
                        const float* __restrict__ b1, const float* __restrict__ w2,
                        const float* __restrict__ b2, float* __restrict__ att) {
  int b = blockIdx.x;
  int t = threadIdx.x;  // 32 threads
  __shared__ float a1[16];
  if (t < 16) {
    float s = b1[t];
    for (int c = 0; c < 32; ++c) s += w1[t * 32 + c] * a0[b * 32 + c];
    a1[t] = fmaxf(s, 0.f);
  }
  __syncthreads();
  float s = b2[t];
  for (int r = 0; r < 16; ++r) s += w2[t * 16 + r] * a1[r];
  att[b * 32 + t] = 1.f / (1.f + __expf(-s));
}

__global__ void k_cab_apply(float* __restrict__ xl, const float* __restrict__ skip2,
                            const float* __restrict__ h, const float* __restrict__ att,
                            long total) {
  long idx = (long)blockIdx.x * blockDim.x + threadIdx.x;
  if (idx >= total) return;
  int c = idx & 31;
  long s = (idx >> 5) & 4095;
  int b = (int)(idx >> 17);
  xl[idx] = xl[idx] * skip2[c] + h[((long)b * 32 + c) * 4096 + s] * att[b * 32 + c];
}

// ---------------------------------------------------------------------------
extern "C" void kernel_launch(void* const* d_in, const int* in_sizes, int n_in,
                              void* d_out, int out_size, void* d_ws, size_t ws_size,
                              hipStream_t stream) {
  (void)in_sizes; (void)n_in; (void)out_size; (void)ws_size;
  auto F = [&](int i) { return (const float*)d_in[i]; };
  const float* X = F(0);
  float* OUT = (float*)d_out;

  float* p = (float*)d_ws;
  auto alloc = [&](long nfl) { float* r = p; p += nfl; return r; };
  float* t_big  = alloc(4L * 32768 * 32);   // expand out (NDHWC); aliased as xs
  float* t_a    = alloc(524288);
  float* t_b    = alloc(524288);
  float* t_c    = alloc(524288);
  float* t_xl   = alloc(524288);
  float* t_h    = alloc(524288);
  float* t_xz   = alloc(1048576);
  float* t_xc   = alloc(524288);
  float* t_xdbl = alloc(4L * 8 * 34 * 4096);
  float* t_delta= alloc(4L * 8 * 32 * 4096);
  float* t_ys   = alloc(4L * 8 * 32 * 4096);
  float* t_ycl  = alloc(524288);
  float* t_ccf  = alloc(524288);
  float* t_cab1 = alloc(4L * 10 * 4096);
  float* t_cab2 = alloc(524288);
  float* t_pool = alloc(128);
  float* t_att  = alloc(128);
  _Float16* wf_down = (_Float16*)alloc(13824);
  _Float16* wf_c1   = (_Float16*)alloc(13824);
  _Float16* wf_c2   = (_Float16*)alloc(13824);
  float* t_xs = t_big;   // alias: expand result dead before xs is built

  const long TOT_CL = 4L * 4096 * 32, TOT_KS = 4L * 8 * 32 * 4096;
  const int EB = 256;
  auto nb = [&](long n) { return (int)((n + EB - 1) / EB); };

  // weight repacks for the implicit-GEMM convs
  k_repack_w<<<108, 256, 0, stream>>>(F(3),  wf_down);
  k_repack_w<<<108, 256, 0, stream>>>(F(9),  wf_c1);
  k_repack_w<<<108, 256, 0, stream>>>(F(13), wf_c2);

  // expand conv 1->32 (NDHWC)
  k_expand<<<nb(4L * 32768 * 32), EB, 0, stream>>>(X, F(1), F(2), t_big, 4L * 32768 * 32);
  // down conv stride2 (WMMA) -> t_a ; GN0+ReLU -> t_b
  k_conv32_wmma<<<4 * 16 * 16, 32, 0, stream>>>(t_big, wf_down, F(4), t_a,
                                                32, 32, 32, 16, 16, 16, 2);
  k_gn_cl<<<32, 256, 0, stream>>>(t_a, t_b, F(5), F(6), 4096, 1);
  // residual conv block
  k_gn_cl<<<32, 256, 0, stream>>>(t_b, t_c, F(7), F(8), 4096, 1);
  k_conv32_wmma<<<1024, 32, 0, stream>>>(t_c, wf_c1, F(10), t_a, 16, 16, 16, 16, 16, 16, 1);
  k_gn_cl<<<32, 256, 0, stream>>>(t_a, t_c, F(11), F(12), 4096, 1);
  k_conv32_wmma<<<1024, 32, 0, stream>>>(t_c, wf_c2, F(14), t_a, 16, 16, 16, 16, 16, 16, 1);
  // 1x1 id conv (WMMA GEMM) + conv-branch residual -> xl (chan-last)
  k_gemm32<false, true><<<256 * 2, 32, 0, stream>>>(t_b, F(15), F(16), t_a, nullptr,
      t_xl, 32,
      0, 4096L * 32, 32, 32,
      0, 4096L * 32, 32, 1,
      0, 4096L * 32, 32);
  k_cl_to_cf<<<nb(TOT_CL), EB, 0, stream>>>(t_xl, OUT, 160, 0, TOT_CL);  // skip 0

  for (int blk = 0; blk < 8; ++blk) {
    int pb = 17 + 25 * blk;
    // ln1 (eps 1e-6)
    k_ln32<<<2048, 256, 0, stream>>>(t_xl, t_h, F(pb + 0), F(pb + 1), 1e-6f,
                                     nullptr, 0, 0, 4096, 16384);
    // in_proj (p,32)x(64,32)^T -> xz (p,64)
    k_gemm32<false, false><<<256 * 4, 32, 0, stream>>>(t_h, F(pb + 6), nullptr,
        nullptr, nullptr, t_xz, 64,
        0, 4096L * 32, 32, 32,
        0, 4096L * 64, 64, 1,
        0, 0, 0);
    // depthwise conv + SiLU straight out of xz -> xc (chan-last)
    k_dwconv_silu<<<nb(TOT_CL), EB, 0, stream>>>(t_xz, F(pb + 7), F(pb + 8), t_xc, TOT_CL);
    // 8 direction sequences, chan-last per direction
    k_gather<<<nb(TOT_KS), EB, 0, stream>>>(t_xc, t_xs, TOT_KS);
    // x_proj per direction: (p,32)x(34,32)^T -> x_dbl[b,k,34,l]
    for (int k = 0; k < 8; ++k) {
      k_gemm32<true, false><<<256 * 3, 32, 0, stream>>>(t_xs,
          F(pb + 9) + (long)k * 34 * 32, nullptr, nullptr, nullptr, t_xdbl, 34,
          (long)k * 4096 * 32, 8L * 4096 * 32, 32, 32,
          (long)k * 34 * 4096, 8L * 34 * 4096, 1, 4096,
          0, 0, 0);
    }
    k_dt<<<nb(TOT_KS), EB, 0, stream>>>(t_xdbl, F(pb + 10), F(pb + 11), t_delta, TOT_KS);
    k_scan<<<32, 512, 0, stream>>>(t_delta, t_xs, t_xdbl, F(pb + 12), F(pb + 13), t_ys);
    k_merge<<<nb(TOT_CL), EB, 0, stream>>>(t_ys, t_ycl, TOT_CL);
    // out-norm fused * silu(z)
    k_ln32<<<2048, 256, 0, stream>>>(t_ycl, t_h, F(pb + 14), F(pb + 15), 1e-5f,
                                     t_xz + 32, 64, 0, 4096, 16384);
    // out proj + xl*skip1 residual -> xl (in place)
    k_gemm32<false, true><<<256 * 2, 32, 0, stream>>>(t_h, F(pb + 16), nullptr,
        t_xl, F(pb + 4), t_xl, 32,
        0, 4096L * 32, 32, 32,
        0, 4096L * 32, 32, 1,
        0, 4096L * 32, 32);
    // ln2 with fused transpose to channel-first for CAB
    k_ln32<<<2048, 256, 0, stream>>>(t_xl, t_ccf, F(pb + 2), F(pb + 3), 1e-5f,
                                     nullptr, 0, 1, 4096, 16384);
    long tot1 = 4L * 10 * 4096;
    k_conv_direct<<<nb(tot1), EB, 0, stream>>>(t_ccf, F(pb + 17), F(pb + 18), t_cab1,
                                               32, 10, 3, tot1);
    k_conv_direct<<<nb(TOT_CL), EB, 0, stream>>>(t_cab1, F(pb + 19), F(pb + 20), t_cab2,
                                                 10, 32, 0, TOT_CL);
    k_pool<<<128, 256, 0, stream>>>(t_cab2, t_pool, 4096);
    k_camlp<<<4, 32, 0, stream>>>(t_pool, F(pb + 21), F(pb + 22), F(pb + 23),
                                  F(pb + 24), t_att);
    k_cab_apply<<<nb(TOT_CL), EB, 0, stream>>>(t_xl, F(pb + 5), t_cab2, t_att, TOT_CL);
    if (blk & 1) {
      k_cl_to_cf<<<nb(TOT_CL), EB, 0, stream>>>(t_xl, OUT, 160,
                                                32 * ((blk >> 1) + 1), TOT_CL);
    }
  }
}
